// GC_GRU_82858509075097
// MI455X (gfx1250) — matrix operand order repaired and checked by reading, not compiled, not run on hardware
//
#include <hip/hip_runtime.h>
#include <hip/hip_bf16.h>

typedef __attribute__((ext_vector_type(16))) _Float16 v16h;
typedef __attribute__((ext_vector_type(8)))  _Float16 v8h;
typedef __attribute__((ext_vector_type(8)))  float    v8f;

#define B_    128
#define N_    512
#define F_    16
#define H_    128
#define HIST_ 24
#define FW_   24
#define E_    8192
#define BN_   (B_*N_)      /* 65536  */
#define ETOT_ (B_*E_)      /* 1048576 */

__device__ __forceinline__ float sigmoidf_(float x){ return 1.0f/(1.0f + __expf(-x)); }

// ---------------------------------------------------------------------------
// One-time setup kernels
// ---------------------------------------------------------------------------
__global__ void k_deg(const int* __restrict__ dst, float* __restrict__ deg){
  int e = blockIdx.x*blockDim.x + threadIdx.x;
  if (e < ETOT_) atomicAdd(&deg[dst[e]], 1.0f);
}

__global__ void k_invdeg(float* __restrict__ deg){
  int g = blockIdx.x*blockDim.x + threadIdx.x;
  if (g < BN_) deg[g] = 1.0f / fmaxf(deg[g], 1.0f);
}

__global__ void k_init_xn(const float* __restrict__ pm25, float* __restrict__ xn){
  int g = blockIdx.x*blockDim.x + threadIdx.x;
  if (g >= BN_) return;
  int b = g >> 9, n = g & (N_-1);
  xn[g] = pm25[((size_t)b*HIST_ + (HIST_-1))*N_ + n];
}

// W_hh (384x128) -> f16 B-fragment layout [ntile(24)][ktile(4)][lane(32)][h(16)]
// B element (k,n): lane = n + 16*((k>>3)&1); h = (k&7) + 8*(k>>4).
__global__ void k_conv_whh(const float* __restrict__ W_hh, _Float16* __restrict__ out){
  int idx = blockIdx.x*blockDim.x + threadIdx.x;
  if (idx >= 24*4*32*16) return;
  int h    = idx & 15;
  int lane = (idx >> 4) & 31;
  int kt   = (idx >> 9) & 3;
  int nt   = idx >> 11;
  int n    = lane & 15;
  int k    = kt*32 + (h < 8 ? h : h + 8) + ((lane >> 4) ? 8 : 0);
  out[idx] = (_Float16)W_hh[(size_t)(nt*16 + n)*H_ + k];
}

// W_ih (384x18) -> f16 B-fragment layout, K padded 18 -> 32: [gtile(24)][lane(32)][h(16)]
__global__ void k_conv_wih(const float* __restrict__ W_ih, _Float16* __restrict__ out){
  int idx = blockIdx.x*blockDim.x + threadIdx.x;
  if (idx >= 24*32*16) return;
  int h    = idx & 15;
  int lane = (idx >> 4) & 31;
  int g    = idx >> 9;
  int n    = lane & 15;
  int k    = (h < 8 ? h : h + 8) + ((lane >> 4) ? 8 : 0);
  out[idx] = (k < (F_+2)) ? (_Float16)W_ih[(size_t)(g*16 + n)*(F_+2) + k] : (_Float16)0.0f;
}

// ---------------------------------------------------------------------------
// Per-step: edge gather/scatter-add  agg[dst] += [xn[src], feat_t[src]]
// ---------------------------------------------------------------------------
__global__ void k_scatter(const int* __restrict__ src, const int* __restrict__ dst,
                          const float* __restrict__ feat, const float* __restrict__ xn,
                          float* __restrict__ agg, int t){
  int e = blockIdx.x*blockDim.x + threadIdx.x;
  if (e >= ETOT_) return;
  int s = src[e], d = dst[e];
  int b = s >> 9, n = s & (N_-1);
  const float* fp = feat + (((size_t)b*(HIST_+FW_) + (HIST_+t))*N_ + n)*F_;
  float* ag = agg + (size_t)d*17;
  atomicAdd(ag, xn[s]);
  #pragma unroll
  for (int f=0; f<F_; ++f) atomicAdd(ag + 1 + f, fp[f]);
}

// ---------------------------------------------------------------------------
// GRU cell with WMMA.  One wave handles 16 nodes.
// ---------------------------------------------------------------------------
__device__ __forceinline__ v8f wmma_step(v16h a, const _Float16* __restrict__ btile,
                                         int lane, v8f c){
  v16h b = *(const v16h*)(btile + (lane << 4));
  return __builtin_amdgcn_wmma_f32_16x16x32_f16(false, a, false, b, (short)0, c,
                                                false, false);
}

__global__ void __launch_bounds__(256, 1) k_gru(
    const _Float16* __restrict__ h16_in, _Float16* __restrict__ h16_out,
    float* __restrict__ hnf,
    const float* __restrict__ xn, const float* __restrict__ agg,
    const float* __restrict__ invdeg, const float* __restrict__ feat,
    const _Float16* __restrict__ whhF, const _Float16* __restrict__ wihF,
    const float* __restrict__ W_nb, const float* __restrict__ b_nb,
    const float* __restrict__ W_root,
    const float* __restrict__ b_ih, const float* __restrict__ b_hh, int t)
{
  const int lane = threadIdx.x & 31;
  const int wave = (blockIdx.x * blockDim.x + threadIdx.x) >> 5;
  const int m0   = wave << 4;
  const int hs   = lane >> 4;          // which K-half this lane supplies
  const int ln   = lane & 15;
  const int row  = m0 + ln;            // node whose A-row this lane carries

  // ---- per-row x2 = [xn, feat(16), gcn], plus inline GCN gate ----
  float x2v[F_+2];
  x2v[0] = xn[row];
  const int bb = row >> 9, nn = row & (N_-1);
  const float* fp = feat + (((size_t)bb*(HIST_+FW_) + (HIST_+t))*N_ + nn)*F_;
  #pragma unroll
  for (int f=0; f<F_; ++f) x2v[1+f] = fp[f];
  const float  idg = invdeg[row];
  const float* ap  = agg + (size_t)row*17;
  float sg = b_nb[0];
  #pragma unroll
  for (int i=0; i<17; ++i) sg += ap[i]*idg*W_nb[i] + x2v[i]*W_root[i];
  x2v[F_+1] = sigmoidf_(sg);

  // A-fragment for gi (K padded 18 -> 32).  A layout (16-bit, 16x32):
  // lanes 0-15: halves 0-7 -> K 0-7, 8-15 -> K 16-23; lanes 16-31: +8.
  v16h xf;
  #pragma unroll
  for (int h=0; h<16; ++h){
    int k = (h < 8 ? h : h + 8) + (hs ? 8 : 0);
    xf[h] = (k < (F_+2)) ? (_Float16)x2v[k] : (_Float16)0.0f;
  }

  // A-fragments for gh: 4 K-tiles of the hidden row (two 16B chunks per lane)
  const _Float16* hrow = h16_in + (size_t)row*H_;
  v16h af[4];
  #pragma unroll
  for (int kt=0; kt<4; ++kt){
    const _Float16* p = hrow + kt*32 + hs*8;
    v8h lo = *(const v8h*)p;
    v8h hi = *(const v8h*)(p + 16);
    af[kt] = __builtin_shufflevector(lo, hi, 0,1,2,3,4,5,6,7,8,9,10,11,12,13,14,15);
  }

  #pragma unroll 2
  for (int j=0; j<8; ++j){
    const int col = j*16 + ln;                  // hidden column 0..127
    const float br  = b_ih[col]        + b_hh[col];
    const float bz  = b_ih[col +   H_] + b_hh[col +   H_];
    const float bni = b_ih[col + 2*H_];
    const float bnh = b_hh[col + 2*H_];
    v8f cr, cz, cni, cnh;
    #pragma unroll
    for (int v=0; v<8; ++v){ cr[v]=br; cz[v]=bz; cni[v]=bni; cnh[v]=bnh; }

    // input-gate contributions (one K=32 WMMA each; K padded with zeros)
    cr  = wmma_step(xf, wihF + (size_t)(j     )*512, lane, cr);
    cz  = wmma_step(xf, wihF + (size_t)(j +  8)*512, lane, cz);
    cni = wmma_step(xf, wihF + (size_t)(j + 16)*512, lane, cni);

    // recurrent contributions: 4 K-tiles over H=128 (n-gate kept separate:
    // GRU applies r BEFORE adding the recurrent part)
    #pragma unroll
    for (int kt=0; kt<4; ++kt){
      cr  = wmma_step(af[kt], whhF + (size_t)((j     )*4 + kt)*512, lane, cr);
      cz  = wmma_step(af[kt], whhF + (size_t)((j +  8)*4 + kt)*512, lane, cz);
      cnh = wmma_step(af[kt], whhF + (size_t)((j + 16)*4 + kt)*512, lane, cnh);
    }

    // element-wise GRU update; C/D layout: lane -> N, vgpr v -> M (+8 for hi lanes)
    #pragma unroll
    for (int v=0; v<8; ++v){
      const int    rr  = m0 + v + hs*8;
      const size_t off = (size_t)rr*H_ + col;
      float r    = sigmoidf_(cr[v]);
      float z    = sigmoidf_(cz[v]);
      float nv   = tanhf(cni[v] + r*cnh[v]);
      float hnew = (1.0f - z)*nv + z*hnf[off];
      hnf[off]     = hnew;
      h16_out[off] = (_Float16)hnew;
    }
  }
}

// ---------------------------------------------------------------------------
// Readout: xn_new = hn @ W_out.T + b_out ; also writes prediction slice t
// ---------------------------------------------------------------------------
__global__ void k_out(const float* __restrict__ hnf, const float* __restrict__ W_out,
                      const float* __restrict__ b_out, float* __restrict__ xn,
                      float* __restrict__ out, int t){
  int g = blockIdx.x*blockDim.x + threadIdx.x;
  if (g >= BN_) return;
  const float4* hp = (const float4*)(hnf + (size_t)g*H_);
  const float4* wp = (const float4*)W_out;
  float s = b_out[0];
  #pragma unroll
  for (int k=0; k<H_/4; ++k){
    float4 h = hp[k], w = wp[k];
    s += h.x*w.x + h.y*w.y + h.z*w.z + h.w*w.w;
  }
  xn[g] = s;
  int b = g >> 9, n = g & (N_-1);
  out[((size_t)b*FW_ + t)*N_ + n] = s;
}

// ---------------------------------------------------------------------------
extern "C" void kernel_launch(void* const* d_in, const int* in_sizes, int n_in,
                              void* d_out, int out_size, void* d_ws, size_t ws_size,
                              hipStream_t stream) {
  (void)in_sizes; (void)n_in; (void)out_size; (void)ws_size;
  const float* feature = (const float*)d_in[0];
  const float* pm25    = (const float*)d_in[1];
  const int*   eidx    = (const int*)  d_in[2];
  const float* W_nb    = (const float*)d_in[3];
  const float* b_nb    = (const float*)d_in[4];
  const float* W_root  = (const float*)d_in[5];
  const float* W_ih    = (const float*)d_in[6];
  const float* W_hh    = (const float*)d_in[7];
  const float* b_ih    = (const float*)d_in[8];
  const float* b_hh    = (const float*)d_in[9];
  const float* W_out   = (const float*)d_in[10];
  const float* b_out   = (const float*)d_in[11];
  float* out = (float*)d_out;

  const int* src = eidx;
  const int* dst = eidx + ETOT_;

  // workspace carve-up (all offsets 256B aligned)
  unsigned char* w = (unsigned char*)d_ws;
  size_t off = 0;
  float*    hnf   = (float*)   (w + off); off += (size_t)BN_*H_*4;   // 32 MB
  _Float16* h16a  = (_Float16*)(w + off); off += (size_t)BN_*H_*2;   // 16 MB
  _Float16* h16b  = (_Float16*)(w + off); off += (size_t)BN_*H_*2;   // 16 MB
  float*    xn    = (float*)   (w + off); off += (size_t)BN_*4;
  float*    agg   = (float*)   (w + off); off += (size_t)BN_*17*4;
  float*    ideg  = (float*)   (w + off); off += (size_t)BN_*4;
  _Float16* whhF  = (_Float16*)(w + off); off += (size_t)24*4*32*16*2;
  _Float16* wihF  = (_Float16*)(w + off); off += (size_t)24*32*16*2;

  // ---- one-time setup ----
  hipMemsetAsync(hnf,  0, (size_t)BN_*H_*4, stream);
  hipMemsetAsync(h16a, 0, (size_t)BN_*H_*2, stream);
  hipMemsetAsync(ideg, 0, (size_t)BN_*4,    stream);

  k_deg     <<<(ETOT_+255)/256, 256, 0, stream>>>(dst, ideg);
  k_invdeg  <<<(BN_  +255)/256, 256, 0, stream>>>(ideg);
  k_init_xn <<<(BN_  +255)/256, 256, 0, stream>>>(pm25, xn);
  k_conv_whh<<<(24*4*32*16+255)/256, 256, 0, stream>>>(W_hh, whhF);
  k_conv_wih<<<(24*32*16  +255)/256, 256, 0, stream>>>(W_ih, wihF);

  // ---- 24-step scan ----
  const int gru_blocks = (BN_/16) / 8;   // 8 waves per 256-thread block
  for (int t = 0; t < FW_; ++t) {
    hipMemsetAsync(agg, 0, (size_t)BN_*17*4, stream);
    k_scatter<<<(ETOT_+255)/256, 256, 0, stream>>>(src, dst, feature, xn, agg, t);

    const _Float16* hin  = (t & 1) ? h16b : h16a;
    _Float16*       hout = (t & 1) ? h16a : h16b;
    k_gru<<<gru_blocks, 256, 0, stream>>>(hin, hout, hnf, xn, agg, ideg, feature,
                                          whhF, wihF, W_nb, b_nb, W_root,
                                          b_ih, b_hh, t);

    k_out<<<(BN_+255)/256, 256, 0, stream>>>(hnf, W_out, b_out, xn, out, t);
  }
}
